// CharRNN_6734508720386
// MI455X (gfx1250) — matrix operand order repaired and checked by reading, not compiled, not run on hardware
//
#include <hip/hip_runtime.h>

// ---------------------------------------------------------------------------
// CharRNN (3-layer LSTM, H=1024, B=512, T=128, V=30) for MI455X / gfx1250.
// bf16 operands + f32 accumulation via v_wmma_f32_16x16x32_bf16.
// Global->LDS staging uses CDNA5 async copies (ASYNCcnt) with double buffering.
// ---------------------------------------------------------------------------

typedef __bf16 bf16_t;
typedef __attribute__((ext_vector_type(16))) __bf16        v16bf;
typedef __attribute__((ext_vector_type(8)))  float         v8f;
typedef __attribute__((ext_vector_type(4)))  unsigned int  u32x4;

#define RNN_B 512
#define RNN_T 128
#define RNN_H 1024
#define RNN_V 30
#define N4H   4096   // 4*H

// ---- bf16 <-> f32 helpers (bit-exact RNE truncation) ----
static __device__ __forceinline__ bf16_t f2bf(float f) {
  union { float f; unsigned u; } a; a.f = f;
  unsigned r = (a.u + 0x7FFFu + ((a.u >> 16) & 1u)) >> 16;
  union { unsigned short s; bf16_t b; } o; o.s = (unsigned short)r;
  return o.b;
}
static __device__ __forceinline__ float bf2f(bf16_t b) {
  union { unsigned short s; bf16_t b; } i; i.b = b;
  union { unsigned u; float f; } o; o.u = ((unsigned)i.s) << 16;
  return o.f;
}
static __device__ __forceinline__ float sigm(float x) {
  return 1.0f / (1.0f + __expf(-x));
}

// Low 32 bits of a generic pointer into __shared__ == LDS byte address
// (ISA 10.2: LDS aperture keeps the LDS offset in addr[31:0]).
static __device__ __forceinline__ unsigned lds_addr32(const void* p) {
  return (unsigned)(unsigned long long)p;
}

// One 16B lane-granular async copy: LDS[lds] = MEM[g..g+15], tracked by ASYNCcnt.
static __device__ __forceinline__ void async_copy_b128(unsigned lds, const bf16_t* g) {
  asm volatile("global_load_async_to_lds_b128 %0, %1, off"
               :: "v"(lds), "v"(g) : "memory");
}

// ---------------------------------------------------------------------------
// f32 -> bf16 weight conversion
// ---------------------------------------------------------------------------
__global__ __launch_bounds__(256) void k_f32_to_bf16(const float* __restrict__ s,
                                                     bf16_t* __restrict__ d, long n) {
  long i = (long)blockIdx.x * 256 + threadIdx.x;
  if (i < n) d[i] = f2bf(s[i]);
}

// ---------------------------------------------------------------------------
// gates[M=512, N=4096] = A1[M,K] * B1[N,K]^T + A2[M,K] * B2[N,K]^T
// Workgroup tile 128x128, 8 waves (2 N-groups x 4 M-groups), wave tile 32x64.
// K-chunks of 32, double-buffered LDS fed by async global->LDS copies.
// ---------------------------------------------------------------------------
__global__ __launch_bounds__(256) void k_gates_gemm(
    float* __restrict__ gates,
    const bf16_t* __restrict__ A1, long lda1, const bf16_t* __restrict__ B1,
    const bf16_t* __restrict__ A2, long lda2, const bf16_t* __restrict__ B2,
    int K)
{
  __shared__ __align__(32) bf16_t sA[2][128][32];
  __shared__ __align__(32) bf16_t sB[2][128][32];

  const int tid  = threadIdx.x;
  const int lane = tid & 31;
  const int wave = tid >> 5;
  const int wm   = wave & 3;   // M group: wm*32
  const int wn   = wave >> 2;  // N group: wn*64
  const int m0   = blockIdx.y * 128;
  const int n0   = blockIdx.x * 128;

  // Two fixed 16B chunks per thread for each of A and B tiles (128x32 bf16).
  const int c0  = tid;        // chunk ids 0..511
  const int c1  = tid + 256;
  const int ra0 = c0 >> 2, ka0 = (c0 & 3) * 8;
  const int ra1 = c1 >> 2, ka1 = (c1 & 3) * 8;

  v8f acc[2][4];
#pragma unroll
  for (int i = 0; i < 2; ++i)
#pragma unroll
    for (int j = 0; j < 4; ++j)
      acc[i][j] = (v8f)0.0f;

  const int kb_a = (lane < 16) ? 0 : 16;  // A frag K base (contiguous, matches B)
  const int kb_b = (lane < 16) ? 0 : 16;  // B frag K base
  const int am   = lane & 15;
  const int bn   = lane & 15;

  for (int pass = 0; pass < 2; ++pass) {
    const bf16_t* A  = pass ? A2 : A1;
    const long    ld = pass ? lda2 : lda1;
    const bf16_t* Bw = pass ? B2 : B1;
    if (A == nullptr) continue;

    // Per-thread global base addresses for this pass.
    const bf16_t* gA0 = A  + (long)(m0 + ra0) * ld + ka0;
    const bf16_t* gA1 = A  + (long)(m0 + ra1) * ld + ka1;
    const bf16_t* gB0 = Bw + (long)(n0 + ra0) * (long)K + ka0;
    const bf16_t* gB1 = Bw + (long)(n0 + ra1) * (long)K + ka1;

    // LDS destinations per buffer (byte addresses).
    unsigned lA0[2], lA1[2], lB0[2], lB1[2];
#pragma unroll
    for (int b = 0; b < 2; ++b) {
      lA0[b] = lds_addr32(&sA[b][ra0][ka0]);
      lA1[b] = lds_addr32(&sA[b][ra1][ka1]);
      lB0[b] = lds_addr32(&sB[b][ra0][ka0]);
      lB1[b] = lds_addr32(&sB[b][ra1][ka1]);
    }

    const int nIter = K >> 5;  // K/32

    // Prologue: stage chunk 0 into buffer 0.
    async_copy_b128(lA0[0], gA0);
    async_copy_b128(lA1[0], gA1);
    async_copy_b128(lB0[0], gB0);
    async_copy_b128(lB1[0], gB1);

    for (int i = 0; i < nIter; ++i) {
      const int buf = i & 1;
      if (i + 1 < nIter) {
        const int kc = (i + 1) << 5;
        async_copy_b128(lA0[buf ^ 1], gA0 + kc);
        async_copy_b128(lA1[buf ^ 1], gA1 + kc);
        async_copy_b128(lB0[buf ^ 1], gB0 + kc);
        async_copy_b128(lB1[buf ^ 1], gB1 + kc);
        // Retire exactly the 4 copies of stage i (async loads complete in order).
        asm volatile("s_wait_asynccnt 0x4" ::: "memory");
      } else {
        asm volatile("s_wait_asynccnt 0x0" ::: "memory");
      }
      __syncthreads();  // stage i visible workgroup-wide

      // A fragments: lane holds row m = lane&15; K = kb_a + 0..15 (contiguous)
      v16bf afrag[2];
#pragma unroll
      for (int ii = 0; ii < 2; ++ii) {
        const bf16_t* p = &sA[buf][wm * 32 + ii * 16 + am][kb_a];
        union { u32x4 q[2]; v16bf v; } u;
        u.q[0] = *(const u32x4*)p;
        u.q[1] = *(const u32x4*)(p + 8);
        afrag[ii] = u.v;
      }
      // B fragments: lane holds col n = lane&15; K = kb_b + 0..15 (contiguous)
      v16bf bfrag[4];
#pragma unroll
      for (int j = 0; j < 4; ++j)
        bfrag[j] = *(const v16bf*)&sB[buf][wn * 64 + j * 16 + bn][kb_b];

#pragma unroll
      for (int ii = 0; ii < 2; ++ii)
#pragma unroll
        for (int j = 0; j < 4; ++j)
          acc[ii][j] = __builtin_amdgcn_wmma_f32_16x16x32_bf16(
              false, afrag[ii], false, bfrag[j], (short)0, acc[ii][j], false, false);

      __syncthreads();  // protect this buffer before stage i+2 refills it
    }
  }

  // Epilogue: C layout -> lane n = lane&15, m = r + 8*(lane>=16)
  const int n_l = lane & 15;
  const int mh  = (lane >> 4) * 8;
#pragma unroll
  for (int i = 0; i < 2; ++i)
#pragma unroll
    for (int j = 0; j < 4; ++j)
#pragma unroll
      for (int r = 0; r < 8; ++r) {
        int m = m0 + wm * 32 + i * 16 + mh + r;
        int n = n0 + wn * 64 + j * 16 + n_l;
        gates[(long)m * N4H + n] = acc[i][j][r];
      }
}

// ---------------------------------------------------------------------------
// LSTM cell elementwise: gates += biases (+ one-hot gather for layer 0),
// apply i,f,g,o; length-masked state freeze; write bf16 layer output in place.
// ---------------------------------------------------------------------------
__global__ __launch_bounds__(256) void k_lstm_cell(
    const float* __restrict__ gates,
    const float* __restrict__ b_ih, const float* __restrict__ b_hh,
    const float* __restrict__ w_ih0,          // non-null only for layer 0
    const int* __restrict__ x, const int* __restrict__ lengths, int t,
    float* __restrict__ h_f32, float* __restrict__ c_f32,
    bf16_t* __restrict__ h_bf,
    bf16_t* __restrict__ act_t, long act_stride)  // base = act + t*H
{
  int idx = blockIdx.x * 256 + threadIdx.x;   // 0 .. B*H-1
  int b  = idx >> 10;
  int hh = idx & (RNN_H - 1);

  const float* gr = gates + (long)b * N4H;
  float gi = gr[hh]              + b_ih[hh]              + b_hh[hh];
  float gf = gr[RNN_H + hh]      + b_ih[RNN_H + hh]      + b_hh[RNN_H + hh];
  float gg = gr[2 * RNN_H + hh]  + b_ih[2 * RNN_H + hh]  + b_hh[2 * RNN_H + hh];
  float go = gr[3 * RNN_H + hh]  + b_ih[3 * RNN_H + hh]  + b_hh[3 * RNN_H + hh];

  if (w_ih0 != nullptr) {  // layer 0: one-hot input => column gather of w_ih_0
    int tok = x[b * RNN_T + t];
    gi += w_ih0[(long)(0 * RNN_H + hh) * RNN_V + tok];
    gf += w_ih0[(long)(1 * RNN_H + hh) * RNN_V + tok];
    gg += w_ih0[(long)(2 * RNN_H + hh) * RNN_V + tok];
    go += w_ih0[(long)(3 * RNN_H + hh) * RNN_V + tok];
  }

  float c_old = c_f32[idx];
  float cn = sigm(gf) * c_old + sigm(gi) * tanhf(gg);
  float hn = sigm(go) * tanhf(cn);

  bool m = (t < lengths[b]);
  if (m) {
    h_f32[idx] = hn;
    c_f32[idx] = cn;
    h_bf[idx]  = f2bf(hn);
  }
  act_t[(long)b * act_stride + hh] = f2bf(m ? hn : 0.0f);
}

// ---------------------------------------------------------------------------
// Copy frozen per-layer final states into d_out (hs / cs regions).
// ---------------------------------------------------------------------------
__global__ __launch_bounds__(256) void k_copy_state(
    const float* __restrict__ h, const float* __restrict__ c,
    float* __restrict__ hs_dst, float* __restrict__ cs_dst)
{
  int i = blockIdx.x * 256 + threadIdx.x;
  hs_dst[i] = h[i];
  cs_dst[i] = c[i];
}

// ---------------------------------------------------------------------------
// Output projection: out[b,t,v] = mask ? dot(act2[b,t,:], w_out[v,:]) + b_out[v] : 0
// One block per (b,t); row staged in LDS; 8 K-slices x 32 v-threads, LDS reduce.
// ---------------------------------------------------------------------------
__global__ __launch_bounds__(256) void k_out_proj(
    const bf16_t* __restrict__ act, const float* __restrict__ w_out,
    const float* __restrict__ b_out, const int* __restrict__ lengths,
    float* __restrict__ out)
{
  __shared__ float row[RNN_H];
  __shared__ float partial[8][32];

  int bt = blockIdx.x;
  int b  = bt >> 7;           // T = 128
  int t  = bt & (RNN_T - 1);

  const bf16_t* a = act + (long)bt * RNN_H;
  for (int i = threadIdx.x; i < RNN_H; i += 256) row[i] = bf2f(a[i]);
  __syncthreads();

  int v  = threadIdx.x & 31;
  int sl = threadIdx.x >> 5;   // 8 slices of 128 K each
  float s = 0.0f;
  if (v < RNN_V) {
    const float* w = w_out + (long)v * RNN_H + sl * 128;
    const float* r = row + sl * 128;
#pragma unroll 8
    for (int k = 0; k < 128; ++k) s += r[k] * w[k];
  }
  partial[sl][v] = s;
  __syncthreads();

  if (threadIdx.x < 32) {
    float acc = 0.0f;
#pragma unroll
    for (int q = 0; q < 8; ++q) acc += partial[q][threadIdx.x];
    if (threadIdx.x < RNN_V) {
      bool m = (t < lengths[b]);
      out[(long)bt * RNN_V + threadIdx.x] = m ? (acc + b_out[threadIdx.x]) : 0.0f;
    }
  }
}

// ---------------------------------------------------------------------------
// Host orchestration
// ---------------------------------------------------------------------------
extern "C" void kernel_launch(void* const* d_in, const int* in_sizes, int n_in,
                              void* d_out, int out_size, void* d_ws, size_t ws_size,
                              hipStream_t stream) {
  (void)in_sizes; (void)n_in; (void)out_size;

  const int*   x       = (const int*)d_in[0];
  const int*   lengths = (const int*)d_in[1];
  const float* w_ih[3] = {(const float*)d_in[2], (const float*)d_in[6],  (const float*)d_in[10]};
  const float* w_hh[3] = {(const float*)d_in[3], (const float*)d_in[7],  (const float*)d_in[11]};
  const float* b_ih[3] = {(const float*)d_in[4], (const float*)d_in[8],  (const float*)d_in[12]};
  const float* b_hh[3] = {(const float*)d_in[5], (const float*)d_in[9],  (const float*)d_in[13]};
  const float* w_out   = (const float*)d_in[14];
  const float* b_out   = (const float*)d_in[15];

  float* out = (float*)d_out;
  const long BH    = (long)RNN_B * RNN_H;                 // 524288
  const long hsOff = (long)RNN_B * RNN_T * RNN_V;         // 1,966,080
  const long csOff = hsOff + 3 * BH;                      // 3,538,944

  // ---- workspace carve-out (256B aligned) ----
  size_t off = 0;
  auto take = [&](size_t bytes) -> void* {
    off = (off + 255) & ~(size_t)255;
    void* p = (char*)d_ws + off;
    off += bytes;
    return p;
  };
  bf16_t* act    = (bf16_t*)take((size_t)RNN_B * RNN_T * RNN_H * 2);  // 128 MB
  float*  gates  = (float*) take((size_t)RNN_B * N4H * 4);            // 8 MB
  float*  h_f32  = (float*) take((size_t)BH * 4);
  float*  c_f32  = (float*) take((size_t)BH * 4);
  bf16_t* h_bf   = (bf16_t*)take((size_t)BH * 2);
  bf16_t* wih_bf[3] = {nullptr, nullptr, nullptr};
  bf16_t* whh_bf[3];
  for (int l = 1; l < 3; ++l) wih_bf[l] = (bf16_t*)take((size_t)N4H * RNN_H * 2);
  for (int l = 0; l < 3; ++l) whh_bf[l] = (bf16_t*)take((size_t)N4H * RNN_H * 2);
  if (off > ws_size) return;  // workspace too small; nothing we can do

  // ---- one-time weight conversion to bf16 (stays resident in 192MB L2) ----
  const long WN = (long)N4H * RNN_H;  // 4,194,304
  dim3 cgrid((unsigned)((WN + 255) / 256));
  for (int l = 0; l < 3; ++l)
    k_f32_to_bf16<<<cgrid, 256, 0, stream>>>(w_hh[l], whh_bf[l], WN);
  for (int l = 1; l < 3; ++l)
    k_f32_to_bf16<<<cgrid, 256, 0, stream>>>(w_ih[l], wih_bf[l], WN);

  const long actStride = (long)RNN_T * RNN_H;  // element stride between batch rows

  for (int l = 0; l < 3; ++l) {
    hipMemsetAsync(h_f32, 0, BH * 4, stream);
    hipMemsetAsync(c_f32, 0, BH * 4, stream);
    hipMemsetAsync(h_bf,  0, BH * 2, stream);

    for (int t = 0; t < RNN_T; ++t) {
      const bf16_t* A1 = (l == 0) ? nullptr : (act + (long)t * RNN_H);
      const bf16_t* B1 = (l == 0) ? nullptr : wih_bf[l];
      // gates = x_t @ w_ih^T + h @ w_hh^T (layer 0: recurrent term only)
      k_gates_gemm<<<dim3(N4H / 128, RNN_B / 128), 256, 0, stream>>>(
          gates, A1, actStride, B1, h_bf, (long)RNN_H, whh_bf[l], RNN_H);

      k_lstm_cell<<<(unsigned)(BH / 256), 256, 0, stream>>>(
          gates, b_ih[l], b_hh[l], (l == 0) ? w_ih[0] : nullptr,
          x, lengths, t, h_f32, c_f32, h_bf,
          act + (long)t * RNN_H, actStride);
    }

    k_copy_state<<<(unsigned)(BH / 256), 256, 0, stream>>>(
        h_f32, c_f32, out + hsOff + (long)l * BH, out + csOff + (long)l * BH);
  }

  // Final projection over layer-2 activations (already length-masked to zero).
  k_out_proj<<<RNN_B * RNN_T, 256, 0, stream>>>(act, w_out, b_out, lengths, out);
}
